// SequentialS2ViTBlock_78314433675298
// MI455X (gfx1250) — compile-verified, matrix operand order checked by reading, not run on hardware
//
#include <hip/hip_runtime.h>
#include <hip/hip_bf16.h>
#include <math.h>

// ---------- types ----------
typedef __attribute__((ext_vector_type(16))) _Float16 v16h;
typedef __attribute__((ext_vector_type(8)))  _Float16 v8h;
typedef __attribute__((ext_vector_type(4)))  _Float16 v4h;
typedef __attribute__((ext_vector_type(8)))  float    v8f;
typedef __attribute__((ext_vector_type(4)))  float    v4f;
typedef __attribute__((ext_vector_type(4)))  unsigned u32x4;
typedef __attribute__((ext_vector_type(8)))  int      i32x8;
typedef __attribute__((ext_vector_type(4)))  int      i32x4;

union Frag16 { v16h v; v8h h[2]; };

#define NTOK   65536   // B*H*W = 16*64*64
#define CDIM   384
#define QKVDIM 1152
#define FFDIM  1536

#define USE_ASYNC_LDS 1
#define USE_TDM 1

#if USE_TDM && __has_builtin(__builtin_amdgcn_tensor_load_to_lds)
#define HAVE_TDM 1
#else
#define HAVE_TDM 0
#endif

// ---------------------------------------------------------------------------
// CDNA5 async global->LDS copy (ASYNCcnt path) + wait.
// LDS byte address = low 32 bits of the flat pointer (ISA 10.2 aperture rule).
// ---------------------------------------------------------------------------
__device__ __forceinline__ void async_copy_b128(void* lds_dst, const void* gsrc)
{
#if USE_ASYNC_LDS
    unsigned lds = (unsigned)(uintptr_t)lds_dst;
    asm volatile("global_load_async_to_lds_b128 %0, %1, off"
                 :: "v"(lds), "v"(gsrc) : "memory");
#else
    *(v8h*)lds_dst = *(const v8h*)gsrc;
#endif
}

__device__ __forceinline__ void async_wait_all()
{
#if USE_ASYNC_LDS
#if __has_builtin(__builtin_amdgcn_s_wait_asynccnt)
    __builtin_amdgcn_s_wait_asynccnt(0);
#else
    asm volatile("s_wait_asynccnt 0" ::: "memory");
#endif
#endif
}

#if HAVE_TDM
// ---------------------------------------------------------------------------
// TDM 2D tile load: move tile_d1 rows x tile_d0 halves (16-bit elements),
// row stride stride_d0 halves, from global memory into LDS (contiguous).
// D# bitfields per CDNA5 ISA 8.3/8.4. Issued by whole wave (EXEC ignored);
// caller gates to one wave per workgroup. Tracked with TENSORcnt.
// ---------------------------------------------------------------------------
__device__ __forceinline__ void tdm_load_tile16(void* lds_dst, const void* gsrc,
                                                int tile_d0, int tile_d1,
                                                long stride_d0)
{
    unsigned lds = (unsigned)(uintptr_t)lds_dst;
    unsigned long long ga = (unsigned long long)(uintptr_t)gsrc;
    u32x4 g0;
    g0.x = 1u;                                            // count=1 (valid), no gather
    g0.y = lds;                                           // lds_addr
    g0.z = (unsigned)(ga & 0xFFFFFFFFu);                  // global_addr[31:0]
    g0.w = (unsigned)((ga >> 32) & 0x01FFFFFFu)           // global_addr[56:32]
         | (2u << 30);                                    // type = 2 ("image")
    i32x8 g1;
    g1[0] = (1 << 16);                                    // wg_mask=0, data_size=1 (2B)
    g1[1] = (int)(((unsigned)tile_d0 & 0xFFFFu) << 16);   // tensor_dim0[15:0]
    g1[2] = (int)((((unsigned)tile_d0 >> 16) & 0xFFFFu)   // tensor_dim0[31:16]
         | (((unsigned)tile_d1 & 0xFFFFu) << 16));        // tensor_dim1[15:0]
    g1[3] = (int)((((unsigned)tile_d1 >> 16) & 0xFFFFu)   // tensor_dim1[31:16]
         | (((unsigned)tile_d0 & 0xFFFFu) << 16));        // tile_dim0
    g1[4] = (int)((unsigned)tile_d1 & 0xFFFFu);           // tile_dim1 (tile_dim2=0)
    g1[5] = (int)((unsigned long long)stride_d0 & 0xFFFFFFFFu);   // dim0_stride[31:0]
    g1[6] = (int)(((unsigned long long)stride_d0 >> 32) & 0xFFFFu); // dim0_stride[47:32]
    g1[7] = 0;
    i32x4 gz = {0, 0, 0, 0};
#if __clang_major__ >= 23
    i32x8 gz8 = {0, 0, 0, 0, 0, 0, 0, 0};
    __builtin_amdgcn_tensor_load_to_lds(g0, g1, gz, gz, gz8, 0);
#else
    __builtin_amdgcn_tensor_load_to_lds(g0, g1, gz, gz, 0);
#endif
}

__device__ __forceinline__ void tdm_wait_all()
{
#if __has_builtin(__builtin_amdgcn_s_wait_tensorcnt)
    __builtin_amdgcn_s_wait_tensorcnt(0);
#else
    asm volatile("s_wait_tensorcnt 0" ::: "memory");
#endif
}
#endif  // HAVE_TDM

// ---------------------------------------------------------------------------
// 0) Weight prep: Bt[n][k] = (f16) B[k][n]   (coalesced read along n)
// ---------------------------------------------------------------------------
__global__ void transpose_f16_kernel(const float* __restrict__ B,
                                     _Float16* __restrict__ Bt, int K, int N)
{
    size_t idx = (size_t)blockIdx.x * blockDim.x + threadIdx.x;
    if (idx >= (size_t)K * N) return;
    int n = (int)(idx % N);
    int k = (int)(idx / N);
    Bt[(size_t)n * K + k] = (_Float16)B[idx];
}

// ---------------------------------------------------------------------------
// 1) shift2d: fp32 in -> f16 out, 4 channels per thread (group-safe: 96%4==0)
// ---------------------------------------------------------------------------
__global__ void shift2d_kernel(const float* __restrict__ x, _Float16* __restrict__ xs)
{
    size_t idx = (size_t)blockIdx.x * blockDim.x + threadIdx.x;   // over NTOK*96
    if (idx >= (size_t)NTOK * (CDIM / 4)) return;
    int c4 = (int)(idx % 96);
    int c  = c4 * 4;
    size_t t = idx / 96;
    int w = (int)(t % 64); size_t t2 = t / 64;
    int h = (int)(t2 % 64); int b = (int)(t2 / 64);
    int g = c / 96;
    int sh = h, sw = w; bool valid = true;
    if      (g == 0) { sh = h - 1; valid = (h > 0);  }
    else if (g == 1) { sh = h + 1; valid = (h < 63); }
    else if (g == 2) { sw = w - 1; valid = (w > 0);  }
    else             { sw = w + 1; valid = (w < 63); }
    v4h o;
    if (valid) {
        v4f f = *(const v4f*)(x + (((size_t)b * 64 + sh) * 64 + sw) * CDIM + c);
        o.x = (_Float16)f.x; o.y = (_Float16)f.y; o.z = (_Float16)f.z; o.w = (_Float16)f.w;
    } else {
        o.x = (_Float16)0.f; o.y = (_Float16)0.f; o.z = (_Float16)0.f; o.w = (_Float16)0.f;
    }
    *(v4h*)(xs + t * CDIM + c) = o;
}

// ---------------------------------------------------------------------------
// 2) f16 WMMA GEMM: C[M,N] = act( A[M,K] @ Bt[N,K]^T )
//    Tiles staged by the Tensor Data Mover (one descriptor per operand tile,
//    issued by wave 0, TENSORcnt-tracked), double-buffered so the DMA for
//    tile k+1 overlaps the 8 WMMAs of tile k. Async-LDS fallback if no TDM.
//    Block: 256 thr = 8 waves; tile 128x128, K-step 32.
// ---------------------------------------------------------------------------
__global__ __launch_bounds__(256)
void gemm_h_wmma(const _Float16* __restrict__ A, const _Float16* __restrict__ Bt,
                 void* __restrict__ Cout, int M, int N, int K, int act, int out_f16)
{
    __shared__ _Float16 As[2][128][32];
    __shared__ _Float16 Bs[2][128][32];

    const int tid  = threadIdx.x;
    const int lane = tid & 31;
    const int wave = tid >> 5;
    const int wm   = wave & 3;
    const int wn   = wave >> 2;
    const int m0   = blockIdx.y * 128;
    const int n0   = blockIdx.x * 128;

#if !HAVE_TDM
    // staging geometry (fallback): 512 x b128 chunks per tile; 2 per thread.
    const int r0  = tid >> 2;
    const int r1  = r0 + 64;
    const int col = (tid & 3) * 8;
    const _Float16* aG0 = A  + (size_t)(m0 + r0) * K + col;
    const _Float16* aG1 = A  + (size_t)(m0 + r1) * K + col;
    const _Float16* bG0 = Bt + (size_t)(n0 + r0) * K + col;
    const _Float16* bG1 = Bt + (size_t)(n0 + r1) * K + col;
#endif

    v8f acc[2][4];
#pragma unroll
    for (int i = 0; i < 2; i++)
#pragma unroll
        for (int j = 0; j < 4; j++)
#pragma unroll
            for (int e = 0; e < 8; e++) acc[i][j][e] = 0.0f;

    const int nk = K >> 5;

    // prologue: stage tile 0 into buffer 0
#if HAVE_TDM
    if (wave == 0) {
        tdm_load_tile16(&As[0][0][0], A  + (size_t)m0 * K, 32, 128, K);
        tdm_load_tile16(&Bs[0][0][0], Bt + (size_t)n0 * K, 32, 128, K);
    }
#else
    async_copy_b128(&As[0][r0][col], aG0);
    async_copy_b128(&As[0][r1][col], aG1);
    async_copy_b128(&Bs[0][r0][col], bG0);
    async_copy_b128(&Bs[0][r1][col], bG1);
#endif

    const int klo = (lane < 16) ? 0 : 8;

    for (int i = 0; i < nk; i++) {
        const int cur = i & 1;
#if HAVE_TDM
        if (wave == 0) tdm_wait_all();
#else
        async_wait_all();
#endif
        __syncthreads();

        if (i + 1 < nk) {                     // stage next tile (other buffer)
            const int kb = (i + 1) << 5;
            const int nxt = cur ^ 1;
#if HAVE_TDM
            if (wave == 0) {
                tdm_load_tile16(&As[nxt][0][0], A  + (size_t)m0 * K + kb, 32, 128, K);
                tdm_load_tile16(&Bs[nxt][0][0], Bt + (size_t)n0 * K + kb, 32, 128, K);
            }
#else
            async_copy_b128(&As[nxt][r0][col], aG0 + kb);
            async_copy_b128(&As[nxt][r1][col], aG1 + kb);
            async_copy_b128(&Bs[nxt][r0][col], bG0 + kb);
            async_copy_b128(&Bs[nxt][r1][col], bG1 + kb);
#endif
        }

        Frag16 afr[2], bfr[4];
#pragma unroll
        for (int mt = 0; mt < 2; mt++) {
            int row = wm * 32 + mt * 16 + (lane & 15);
            afr[mt].h[0] = *(const v8h*)(&As[cur][row][klo]);
            afr[mt].h[1] = *(const v8h*)(&As[cur][row][klo + 16]);
        }
#pragma unroll
        for (int nt = 0; nt < 4; nt++) {
            int c2 = wn * 64 + nt * 16 + (lane & 15);
            bfr[nt].h[0] = *(const v8h*)(&Bs[cur][c2][klo]);
            bfr[nt].h[1] = *(const v8h*)(&Bs[cur][c2][klo + 16]);
        }
#pragma unroll
        for (int mt = 0; mt < 2; mt++)
#pragma unroll
            for (int nt = 0; nt < 4; nt++)
                acc[mt][nt] = __builtin_amdgcn_wmma_f32_16x16x32_f16(
                    false, afr[mt].v, false, bfr[nt].v,
                    (short)0, acc[mt][nt], false, false);
    }

    // epilogue
    float*    Cf = (float*)Cout;
    _Float16* Ch = (_Float16*)Cout;
#pragma unroll
    for (int mt = 0; mt < 2; mt++)
#pragma unroll
        for (int nt = 0; nt < 4; nt++)
#pragma unroll
            for (int r = 0; r < 8; r++) {
                int row = m0 + wm * 32 + mt * 16 + r + ((lane < 16) ? 0 : 8);
                int c2  = n0 + wn * 64 + nt * 16 + (lane & 15);
                float v = acc[mt][nt][r];
                if (act == 1)
                    v = 0.5f * v * (1.0f + erff(v * 0.70710678118f));
                if (out_f16) Ch[(size_t)row * N + c2] = (_Float16)v;
                else         Cf[(size_t)row * N + c2] = v;
            }
}

// ---------------------------------------------------------------------------
// 3) Windowed MHSA, f16 qkv in / f16 O out. Block per (window, head), 4 waves.
//    Q/K staged via GLOBAL_LOAD_ASYNC_TO_LDS_B128 (ASYNCcnt path).
// ---------------------------------------------------------------------------
__global__ __launch_bounds__(128)
void win_attn_kernel(const _Float16* __restrict__ qkv, _Float16* __restrict__ O)
{
    __shared__ _Float16 Qs[64][32];
    __shared__ _Float16 Ks[64][32];
    __shared__ _Float16 Vt[32][64];
    __shared__ float    Ls[64][64];
    __shared__ _Float16 Ps[64][64];

    const int head = blockIdx.x % 12;
    const int wi   = blockIdx.x / 12;
    const int b    = wi >> 6;
    const int wh   = (wi >> 3) & 7;
    const int ww   = wi & 7;
    const int tid  = threadIdx.x;
    const int lane = tid & 31;
    const int wv   = tid >> 5;

#pragma unroll
    for (int i = 0; i < 2; i++) {
        int ch = tid + i * 128;          // 0..255
        int t  = ch >> 2;                // token 0..63
        int d  = (ch & 3) * 8;           // 0,8,16,24
        size_t rowg = (((size_t)(b * 64 + wh * 8 + (t >> 3)) * 64)
                       + ww * 8 + (t & 7)) * QKVDIM + head * 32 + d;
        async_copy_b128(&Qs[t][d], qkv + rowg);
        async_copy_b128(&Ks[t][d], qkv + rowg + 384);
        v8h vv = *(const v8h*)(qkv + rowg + 768);
#pragma unroll
        for (int j = 0; j < 8; j++) Vt[d + j][t] = vv[j];
    }
    async_wait_all();
    __syncthreads();

    const int klo = (lane < 16) ? 0 : 8;

    // ---- logits ----
    {
        Frag16 afr;
        int row = wv * 16 + (lane & 15);
        afr.h[0] = *(const v8h*)(&Qs[row][klo]);
        afr.h[1] = *(const v8h*)(&Qs[row][klo + 16]);
        const float scale = 0.17677669529663687f;   // 1/sqrt(32)
#pragma unroll
        for (int nt = 0; nt < 4; nt++) {
            Frag16 bfr;
            int c2 = nt * 16 + (lane & 15);
            bfr.h[0] = *(const v8h*)(&Ks[c2][klo]);
            bfr.h[1] = *(const v8h*)(&Ks[c2][klo + 16]);
            v8f acc;
#pragma unroll
            for (int e = 0; e < 8; e++) acc[e] = 0.0f;
            acc = __builtin_amdgcn_wmma_f32_16x16x32_f16(
                false, afr.v, false, bfr.v, (short)0, acc, false, false);
#pragma unroll
            for (int r = 0; r < 8; r++)
                Ls[wv * 16 + r + ((lane < 16) ? 0 : 8)][nt * 16 + (lane & 15)]
                    = acc[r] * scale;
        }
    }
    __syncthreads();

    // ---- softmax per row ----
    if (tid < 64) {
        float mx = -1e30f;
        for (int j = 0; j < 64; j++) mx = fmaxf(mx, Ls[tid][j]);
        float s = 0.0f;
        for (int j = 0; j < 64; j++) { float e = expf(Ls[tid][j] - mx); Ls[tid][j] = e; s += e; }
        float inv = 1.0f / s;
        for (int j = 0; j < 64; j++) Ps[tid][j] = (_Float16)(Ls[tid][j] * inv);
    }
    __syncthreads();

    // ---- O = P @ V ----
    v8f oacc[2];
#pragma unroll
    for (int nt = 0; nt < 2; nt++)
#pragma unroll
        for (int e = 0; e < 8; e++) oacc[nt][e] = 0.0f;

#pragma unroll
    for (int ks = 0; ks < 2; ks++) {
        Frag16 afr;
        int row = wv * 16 + (lane & 15);
        int kh0 = ks * 32 + klo;
        afr.h[0] = *(const v8h*)(&Ps[row][kh0]);
        afr.h[1] = *(const v8h*)(&Ps[row][kh0 + 16]);
#pragma unroll
        for (int nt = 0; nt < 2; nt++) {
            Frag16 bfr;
            int c2 = nt * 16 + (lane & 15);
            bfr.h[0] = *(const v8h*)(&Vt[c2][kh0]);
            bfr.h[1] = *(const v8h*)(&Vt[c2][kh0 + 16]);
            oacc[nt] = __builtin_amdgcn_wmma_f32_16x16x32_f16(
                false, afr.v, false, bfr.v, (short)0, oacc[nt], false, false);
        }
    }

#pragma unroll
    for (int nt = 0; nt < 2; nt++)
#pragma unroll
        for (int r = 0; r < 8; r++) {
            int t = wv * 16 + r + ((lane < 16) ? 0 : 8);
            int d = nt * 16 + (lane & 15);
            size_t rowg = ((size_t)(b * 64 + wh * 8 + (t >> 3)) * 64)
                          + ww * 8 + (t & 7);
            O[rowg * CDIM + head * 32 + d] = (_Float16)oacc[nt][r];
        }
}

// ---------------------------------------------------------------------------
// 4) out = xin + LN(a)*gamma ; optional f16 mirror for the next GEMM's A.
// ---------------------------------------------------------------------------
__global__ __launch_bounds__(128)
void add_ln_kernel(const float* __restrict__ xin, const float* __restrict__ a,
                   const float* __restrict__ gamma, float* __restrict__ out,
                   _Float16* __restrict__ out_h)
{
    __shared__ float rs[128], rq[128];
    const size_t tok = blockIdx.x;
    const int tid = threadIdx.x;
    const size_t base = tok * CDIM;
    float v0 = a[base + tid];
    float v1 = a[base + tid + 128];
    float v2 = a[base + tid + 256];
    rs[tid] = v0 + v1 + v2;
    rq[tid] = v0 * v0 + v1 * v1 + v2 * v2;
    __syncthreads();
    for (int s = 64; s > 0; s >>= 1) {
        if (tid < s) { rs[tid] += rs[tid + s]; rq[tid] += rq[tid + s]; }
        __syncthreads();
    }
    float mean = rs[0] * (1.0f / 384.0f);
    float var  = rq[0] * (1.0f / 384.0f) - mean * mean;
    float inv  = rsqrtf(var + 1e-5f);
    float o0 = xin[base + tid]       + (v0 - mean) * inv * gamma[tid];
    float o1 = xin[base + tid + 128] + (v1 - mean) * inv * gamma[tid + 128];
    float o2 = xin[base + tid + 256] + (v2 - mean) * inv * gamma[tid + 256];
    out[base + tid]       = o0;
    out[base + tid + 128] = o1;
    out[base + tid + 256] = o2;
    if (out_h) {
        out_h[base + tid]       = (_Float16)o0;
        out_h[base + tid + 128] = (_Float16)o1;
        out_h[base + tid + 256] = (_Float16)o2;
    }
}

// ---------------------------------------------------------------------------
// launch
// ---------------------------------------------------------------------------
extern "C" void kernel_launch(void* const* d_in, const int* in_sizes, int n_in,
                              void* d_out, int out_size, void* d_ws, size_t ws_size,
                              hipStream_t stream)
{
    const float* x          = (const float*)d_in[0];
    const float* w_qkv      = (const float*)d_in[1];
    const float* w_out      = (const float*)d_in[2];
    const float* gamma_attn = (const float*)d_in[3];
    const float* w1         = (const float*)d_in[4];
    const float* w2         = (const float*)d_in[5];
    const float* gamma_ff   = (const float*)d_in[6];
    float* out = (float*)d_out;

    const size_t NT = NTOK;
    char* p = (char*)d_ws;
    _Float16* xs_h   = (_Float16*)p; p += NT * CDIM   * sizeof(_Float16);
    _Float16* qkv_h  = (_Float16*)p; p += NT * QKVDIM * sizeof(_Float16);
    _Float16* o_h    = (_Float16*)p; p += NT * CDIM   * sizeof(_Float16);
    float*    a_f    = (float*)p;    p += NT * CDIM   * sizeof(float);
    float*    x1_f   = (float*)p;    p += NT * CDIM   * sizeof(float);
    _Float16* x1_h   = (_Float16*)p; p += NT * CDIM   * sizeof(_Float16);
    _Float16* g_h    = (_Float16*)p; p += NT * FFDIM  * sizeof(_Float16);
    float*    f_f    = (float*)p;    p += NT * CDIM   * sizeof(float);
    _Float16* wqkv_t = (_Float16*)p; p += (size_t)CDIM * QKVDIM * sizeof(_Float16);
    _Float16* wout_t = (_Float16*)p; p += (size_t)CDIM * CDIM   * sizeof(_Float16);
    _Float16* w1_t   = (_Float16*)p; p += (size_t)CDIM * FFDIM  * sizeof(_Float16);
    _Float16* w2_t   = (_Float16*)p; p += (size_t)FFDIM * CDIM  * sizeof(_Float16);

    // 0) weight transpose+convert (one-shot, tiny)
    {
        size_t n;
        n = (size_t)CDIM * QKVDIM;
        transpose_f16_kernel<<<(unsigned)((n + 255) / 256), 256, 0, stream>>>(w_qkv, wqkv_t, CDIM, QKVDIM);
        n = (size_t)CDIM * CDIM;
        transpose_f16_kernel<<<(unsigned)((n + 255) / 256), 256, 0, stream>>>(w_out, wout_t, CDIM, CDIM);
        n = (size_t)CDIM * FFDIM;
        transpose_f16_kernel<<<(unsigned)((n + 255) / 256), 256, 0, stream>>>(w1, w1_t, CDIM, FFDIM);
        n = (size_t)FFDIM * CDIM;
        transpose_f16_kernel<<<(unsigned)((n + 255) / 256), 256, 0, stream>>>(w2, w2_t, FFDIM, CDIM);
    }

    // 1) shift2d -> xs_h (f16)
    {
        size_t n = NT * (CDIM / 4);
        shift2d_kernel<<<(unsigned)((n + 255) / 256), 256, 0, stream>>>(x, xs_h);
    }

    // 2) qkv = xs @ w_qkv  -> f16
    gemm_h_wmma<<<dim3(QKVDIM / 128, NT / 128), 256, 0, stream>>>(
        xs_h, wqkv_t, qkv_h, (int)NT, QKVDIM, CDIM, 0, 1);

    // 3) windowed attention -> o_h (f16)
    win_attn_kernel<<<1024 * 12, 128, 0, stream>>>(qkv_h, o_h);

    // 4) a = o @ w_out -> f32
    gemm_h_wmma<<<dim3(CDIM / 128, NT / 128), 256, 0, stream>>>(
        o_h, wout_t, a_f, (int)NT, CDIM, CDIM, 0, 0);

    // 5) x1 = x + LN(a)*gamma_attn  (f32 + f16 mirror)
    add_ln_kernel<<<(unsigned)NT, 128, 0, stream>>>(x, a_f, gamma_attn, x1_f, x1_h);

    // 6) g = gelu(x1 @ w1) -> f16
    gemm_h_wmma<<<dim3(FFDIM / 128, NT / 128), 256, 0, stream>>>(
        x1_h, w1_t, g_h, (int)NT, FFDIM, CDIM, 1, 1);

    // 7) f = g @ w2 -> f32
    gemm_h_wmma<<<dim3(CDIM / 128, NT / 128), 256, 0, stream>>>(
        g_h, w2_t, f_f, (int)NT, CDIM, FFDIM, 0, 0);

    // 8) out = x1 + LN(f)*gamma_ff
    add_ln_kernel<<<(unsigned)NT, 128, 0, stream>>>(x1_f, f_f, gamma_ff, out, nullptr);
}
